// Attn_11879879543355
// MI455X (gfx1250) — matrix operand order repaired and checked by reading, not compiled, not run on hardware
//
#include <hip/hip_runtime.h>
#include <hip/hip_bf16.h>

// ---------------------------------------------------------------------------
// Types for CDNA5 WMMA (wave32): V_WMMA_F32_16X16X32_BF16
// ---------------------------------------------------------------------------
typedef __bf16 bf16;
typedef bf16  v16bf __attribute__((ext_vector_type(16)));
typedef float v8f   __attribute__((ext_vector_type(8)));

#define WMMA_BF16(a, b, c) \
  __builtin_amdgcn_wmma_f32_16x16x32_bf16(false, (a), false, (b), (short)0, (c), false, false)

// Problem constants (from reference setup_inputs)
#define BATCH  4
#define L_SEQ  2048
#define DMODEL 2048
#define NQKV   6144
#define NH     32
#define DH     64
#define NB     32      // L / WINDOW
#define NSINK  16

// ---------------------------------------------------------------------------
// CDNA5 async global->LDS copies (ASYNCcnt), with sync fallback.
// Builtin signature (probe-confirmed): arg0 = int4 addrspace(1)*,
// arg1 = int4 addrspace(3)*, arg2/arg3 = constant int offset / cpol.
// ---------------------------------------------------------------------------
#if __has_builtin(__builtin_amdgcn_global_load_async_to_lds_b128) && \
    __has_builtin(__builtin_amdgcn_s_wait_asynccnt)
#define USE_ASYNC_LDS 1
#else
#define USE_ASYNC_LDS 0
#endif

typedef int v4i_vs __attribute__((__vector_size__(4 * sizeof(int))));
typedef __attribute__((address_space(1))) v4i_vs as1_v4i;
typedef __attribute__((address_space(3))) v4i_vs as3_v4i;

static __device__ __forceinline__ void cp16(bf16* l, const bf16* g) {
#if USE_ASYNC_LDS
  __builtin_amdgcn_global_load_async_to_lds_b128(
      (as1_v4i*)(g), (as3_v4i*)(l), 0, 0);
#else
  *(uint4*)l = *(const uint4*)g;
#endif
}
#if USE_ASYNC_LDS
#define WAIT_ASYNC(n) __builtin_amdgcn_s_wait_asynccnt(n)
#else
#define WAIT_ASYNC(n) do {} while (0)
#endif

static __device__ __forceinline__ bf16 f2bf(float f) {
  unsigned u = __builtin_bit_cast(unsigned, f);
  unsigned r = (u + 0x7fffu + ((u >> 16) & 1u)) >> 16;   // round-to-nearest-even
  unsigned short s = (unsigned short)r;
  return __builtin_bit_cast(bf16, s);
}
static __device__ __forceinline__ v8f zero8() {
  v8f v;
#pragma unroll
  for (int i = 0; i < 8; ++i) v[i] = 0.0f;
  return v;
}
// A-fragment (16x32 bf16, ISA 7.12.2): lane m = lane&15, kb = (lane>=16)*8.
// elements 0..7 = K [kb, kb+8), elements 8..15 = K [16+kb, 16+kb+8).
static __device__ __forceinline__ v16bf ldfragA(const bf16* p) {
  union { v16bf v; uint4 q[2]; } u;
  u.q[0] = *(const uint4*)(p);
  u.q[1] = *(const uint4*)(p + 16);
  return u.v;
}
// B-fragment (32x16 bf16): lane n = lane&15, kb = (lane>=16)*8;
// elements 0..7 = K [kb, kb+8), 8..15 = K [16+kb, 16+kb+8), stored [n][k] in LDS.
static __device__ __forceinline__ v16bf ldfragB(const bf16* p) {
  union { v16bf v; uint4 q[2]; } u;
  u.q[0] = *(const uint4*)(p);
  u.q[1] = *(const uint4*)(p + 16);
  return u.v;
}

// ---------------------------------------------------------------------------
// Pre-pass: f32 -> bf16, and RoPE cos/sin table
// ---------------------------------------------------------------------------
__global__ __launch_bounds__(256) void cvt_kernel(const float* __restrict__ x,
                                                  bf16* __restrict__ y, int n) {
  int i = blockIdx.x * 256 + threadIdx.x;
  if (i < n) y[i] = f2bf(x[i]);
}

__global__ __launch_bounds__(256) void rope_kernel(float* __restrict__ c,
                                                   float* __restrict__ s) {
  int i = blockIdx.x * 256 + threadIdx.x;   // 2048 * 32
  int p  = i >> 5;
  int hi = i & 31;
  float inv = __powf(10000.0f, -(float)hi / 32.0f);
  float ang = (float)p * inv;
  c[i] = __cosf(ang);
  s[i] = __sinf(ang);
}

// ---------------------------------------------------------------------------
// GEMM core: C(128x128) = A(128xK) * B(128xK)^T, bf16 in, f32 acc.
// 256 threads = 8 waves in a 4x2 grid; each wave owns a 32x64 block = 2x4
// WMMA tiles (8 wmma per 6 fragment loads per 32-K step). LDS rows padded to
// 72 elements (stride 144B = 36 banks) -> conflict-free ds_load_b128 frags.
// Double-buffered tiles staged with CDNA5 async global->LDS copies; the
// double-buffer is selected by an INTEGER offset from the single shared-memory
// base pointer so addrspace(3) inference holds (pointer arrays would lower the
// fragment loads to flat_load, paying LOADcnt+DScnt on the slow path).
// ---------------------------------------------------------------------------
#define LDT 72
#define TSZ (128 * LDT)      // one tile buffer (A or B), elements

static __device__ __forceinline__ void issue_tile_loads(
    const bf16* __restrict__ Ag, const bf16* __restrict__ Bg, int K, int k0,
    bf16* lds, int bufOff) {
  const int tid = threadIdx.x;
#pragma unroll
  for (int i = 0; i < 4; ++i) {              // A tile: 128 x 64 bf16
    int c = tid + i * 256;
    int r = c >> 3, kc = (c & 7) * 8;
    cp16(&lds[bufOff + r * LDT + kc], &Ag[(size_t)r * K + k0 + kc]);
  }
#pragma unroll
  for (int i = 0; i < 4; ++i) {              // B tile: 128 x 64 bf16
    int c = tid + i * 256;
    int r = c >> 3, kc = (c & 7) * 8;
    cp16(&lds[bufOff + TSZ + r * LDT + kc], &Bg[(size_t)r * K + k0 + kc]);
  }
}

static __device__ __forceinline__ void gemm_tile_128x128(
    const bf16* __restrict__ Ag, const bf16* __restrict__ Bg, int K,
    bf16* lds, v8f acc[2][4]) {
  const int tid  = threadIdx.x;
  const int lane = tid & 31, wave = tid >> 5;
  const int wm = wave >> 1, wn = wave & 1;     // 4 x 2 wave grid
  const int m16 = lane >> 4, mrow = lane & 15;
#pragma unroll
  for (int ti = 0; ti < 2; ++ti)
#pragma unroll
    for (int tj = 0; tj < 4; ++tj) acc[ti][tj] = zero8();

  issue_tile_loads(Ag, Bg, K, 0, lds, 0);
  const int iters = K >> 6;
  for (int it = 0; it < iters; ++it) {
    const int cur = (it & 1) * (2 * TSZ);      // integer buffer offset
    const int nxt = ((it & 1) ^ 1) * (2 * TSZ);
    if (it + 1 < iters) {
      // overwrites the buffer last read in the previous iteration (barrier-protected)
      issue_tile_loads(Ag, Bg, K, (it + 1) << 6, lds, nxt);
      WAIT_ASYNC(8);             // leave only the next tile's 8 ops in flight
    } else {
      WAIT_ASYNC(0);             // drain
    }
    __syncthreads();
#pragma unroll
    for (int s = 0; s < 2; ++s) {
      const int ks = s * 32;
      const int aOff = cur + ks + m16 * 8;
      const int bOff = cur + TSZ + ks + m16 * 8;
      v16bf a0 = ldfragA(&lds[aOff + (wm * 32 +      mrow) * LDT]);
      v16bf a1 = ldfragA(&lds[aOff + (wm * 32 + 16 + mrow) * LDT]);
      v16bf b0 = ldfragB(&lds[bOff + (wn * 64 +      mrow) * LDT]);
      v16bf b1 = ldfragB(&lds[bOff + (wn * 64 + 16 + mrow) * LDT]);
      v16bf b2 = ldfragB(&lds[bOff + (wn * 64 + 32 + mrow) * LDT]);
      v16bf b3 = ldfragB(&lds[bOff + (wn * 64 + 48 + mrow) * LDT]);
      acc[0][0] = WMMA_BF16(a0, b0, acc[0][0]);
      acc[0][1] = WMMA_BF16(a0, b1, acc[0][1]);
      acc[0][2] = WMMA_BF16(a0, b2, acc[0][2]);
      acc[0][3] = WMMA_BF16(a0, b3, acc[0][3]);
      acc[1][0] = WMMA_BF16(a1, b0, acc[1][0]);
      acc[1][1] = WMMA_BF16(a1, b1, acc[1][1]);
      acc[1][2] = WMMA_BF16(a1, b2, acc[1][2]);
      acc[1][3] = WMMA_BF16(a1, b3, acc[1][3]);
    }
    __syncthreads();   // everyone done reading `cur` before it is refilled
  }
}

#define GEMM_SMEM (4 * TSZ * 2)        // 73728 B (also covers 128x128 f32 C)

// ---------------------------------------------------------------------------
// Kernel 1: qkv = u @ Wqkv^T + bqkv, fused RoPE on q/k, store head-major bf16.
// blockIdx.x: 48 N-tiles of 128 cols (= two heads, never crossing q/k/v).
// blockIdx.y: 64 M-tiles of 128 rows.
// ---------------------------------------------------------------------------
__global__ __launch_bounds__(256) void qkv_rope_kernel(
    const bf16* __restrict__ U, const bf16* __restrict__ W,
    const float* __restrict__ bias,
    const float* __restrict__ ropeC, const float* __restrict__ ropeS,
    bf16* __restrict__ Qw, bf16* __restrict__ Kw, bf16* __restrict__ Vw) {
  extern __shared__ char smem[];
  v8f acc[2][4];
  gemm_tile_128x128(U + (size_t)blockIdx.y * 128 * DMODEL,
                    W + (size_t)blockIdx.x * 128 * DMODEL, DMODEL,
                    (bf16*)smem, acc);

  float* Cs = (float*)smem;            // 128 x 128 f32 (64 KB), reuses tiles
  const int tid  = threadIdx.x;
  const int lane = tid & 31, wave = tid >> 5;
  const int wm = wave >> 1, wn = wave & 1;
  const int m16 = lane >> 4, mrow = lane & 15;
#pragma unroll
  for (int ti = 0; ti < 2; ++ti)
#pragma unroll
    for (int tj = 0; tj < 4; ++tj)
#pragma unroll
      for (int r = 0; r < 8; ++r) {
        int row = wm * 32 + ti * 16 + r + m16 * 8;
        int col = wn * 64 + tj * 16 + mrow;
        Cs[row * 128 + col] = acc[ti][tj][r];
      }
  __syncthreads();

  const int gcol0 = blockIdx.x * 128;
  const int mat   = gcol0 >> 11;          // 0=q 1=k 2=v (constant per block)
  bf16* dst = (mat == 0) ? Qw : (mat == 1) ? Kw : Vw;
#pragma unroll 4
  for (int i = 0; i < 64; ++i) {
    int c = tid + i * 256;                // 0..16383 over the 128x128 tile
    int r = c >> 7, cc = c & 127;
    int g = blockIdx.y * 128 + r;
    int b = g >> 11, l = g & 2047;
    int gcol = gcol0 + cc;
    int h  = (gcol & 2047) >> 6;
    int dd = cc & 63;
    float x = Cs[r * 128 + cc] + bias[gcol];
    float val = x;
    if (mat < 2) {                        // RoPE pairs dd with dd^32 (same head)
      int hi = dd & 31;
      float cs = ropeC[l * 32 + hi], sn = ropeS[l * 32 + hi];
      float other = Cs[r * 128 + (cc ^ 32)] + bias[gcol ^ 32];
      val = (dd < 32) ? (x * cs - other * sn) : (other * sn + x * cs);
    }
    dst[(((size_t)b * NH + h) * L_SEQ + l) * DH + dd] = f2bf(val);
  }
}

// ---------------------------------------------------------------------------
// Kernel 2: banded attention + sinks. One workgroup (128 thr = 4 waves) per
// (b, h, block). Scores 64 x (16 sink + 192 band) via WMMA (K=64), masked
// softmax in LDS, then O = P @ [sinks;Vband;0] via WMMA with K padded to 224.
// ---------------------------------------------------------------------------
#define NKEY  208                 // 16 sinks + 192 band keys
#define KPAD  224                 // NKEY rounded up to 32
#define SROW  212                 // f32 score row stride
#define PROW  232                 // bf16 P / SVt row stride (16B aligned)

__global__ __launch_bounds__(128) void attn_kernel(
    const bf16* __restrict__ Q, const bf16* __restrict__ K,
    const bf16* __restrict__ V, const bf16* __restrict__ Snk,
    bf16* __restrict__ O) {
  extern __shared__ char smem[];
  bf16*  Qs  = (bf16*)smem;                 // 64  x 72
  bf16*  KS  = Qs  + 64 * LDT;              // 208 x 72   [key][channel]
  bf16*  SVt = KS  + NKEY * LDT;            // 64  x 232  [channel][key]
  bf16*  P   = SVt + 64 * PROW;             // 64  x 232  probabilities
  float* S   = (float*)(P + 64 * PROW);     // 64  x 212  raw scores

  const int bid = blockIdx.x;
  const int bi = bid & (NB - 1);
  const int h  = (bid >> 5) & (NH - 1);
  const int b  = bid >> 10;
  const int tid = threadIdx.x;
  const bf16 bz = __builtin_bit_cast(bf16, (unsigned short)0);

  const size_t headoff = (((size_t)b * NH + h) * L_SEQ) * DH;
  const bf16* Qh = Q + headoff;
  const bf16* Kh = K + headoff;
  const bf16* Vh = V + headoff;
  const bf16* Sh = Snk + (size_t)h * NSINK * DH;

  // --- stage Q (64x64) ---
#pragma unroll
  for (int i = 0; i < 4; ++i) {
    int c = tid + i * 128;
    int r = c >> 3, kc = (c & 7) * 8;
    *(uint4*)&Qs[r * LDT + kc] = *(const uint4*)&Qh[(size_t)(bi * 64 + r) * DH + kc];
  }
  // --- stage [sinks ; K band] (208x64), invalid band blocks -> 0 ---
#pragma unroll
  for (int i = 0; i < 13; ++i) {
    int c = tid + i * 128;
    int r = c >> 3, kc = (c & 7) * 8;
    uint4 val = make_uint4(0u, 0u, 0u, 0u);
    if (r < NSINK) {
      val = *(const uint4*)&Sh[r * DH + kc];
    } else {
      int j = r - NSINK;
      int blk = bi - 1 + (j >> 6);
      if (blk >= 0 && blk < NB)
        val = *(const uint4*)&Kh[(size_t)(blk * 64 + (j & 63)) * DH + kc];
    }
    *(uint4*)&KS[r * LDT + kc] = val;
  }
  // --- stage transposed [sinks ; V band ; zero-pad] as SVt[dd][k] ---
  for (int i = 0; i < 112; ++i) {
    int c = tid + i * 128;
    int k = c >> 6, dd = c & 63;
    bf16 v = bz;
    if (k < NSINK) {
      v = Sh[k * DH + dd];
    } else if (k < NKEY) {
      int j = k - NSINK;
      int blk = bi - 1 + (j >> 6);
      if (blk >= 0 && blk < NB) v = Vh[(size_t)(blk * 64 + (j & 63)) * DH + dd];
    }
    SVt[dd * PROW + k] = v;
  }
  __syncthreads();

  const int lane = tid & 31, wave = tid >> 5;
  const int m16 = lane >> 4, mrow = lane & 15;
  const int m0 = wave * 16;                 // each wave owns 16 query rows
  const float scale = 0.125f;               // 1/sqrt(64)

  // --- scores: 13 N-tiles of 16 keys, K-dim = 64 (2 wmma steps) ---
  for (int nt = 0; nt < 13; ++nt) {
    v8f acc = zero8();
#pragma unroll
    for (int s = 0; s < 2; ++s) {
      int ks = s * 32;
      v16bf a  = ldfragA(&Qs[(m0 + mrow) * LDT + ks + m16 * 8]);
      v16bf bb = ldfragB(&KS[(nt * 16 + mrow) * LDT + ks + m16 * 8]);
      acc = WMMA_BF16(a, bb, acc);
    }
#pragma unroll
    for (int r = 0; r < 8; ++r) {
      int row = m0 + r + m16 * 8;
      int col = nt * 16 + mrow;
      float sv = acc[r] * scale;
      if (col >= NSINK) {
        int j = col - NSINK;
        int blk = bi - 1 + (j >> 6);
        int qpos = bi * 64 + row;
        int kpos = blk * 64 + (j & 63);
        int dpq = qpos - kpos; if (dpq < 0) dpq = -dpq;
        bool ok = (blk >= 0) && (blk < NB) && (dpq <= 64);
        if (!ok) sv = -__builtin_inff();
      }
      S[row * SROW + col] = sv;
    }
  }
  __syncthreads();

  // --- masked softmax, one thread per query row ---
  if (tid < 64) {
    float* srow = &S[tid * SROW];
    float mx = -__builtin_inff();
    for (int c2 = 0; c2 < NKEY; ++c2) mx = fmaxf(mx, srow[c2]);
    float sum = 0.0f;
    for (int c2 = 0; c2 < NKEY; ++c2) {
      float e = __expf(srow[c2] - mx);
      srow[c2] = e;
      sum += e;
    }
    float inv = 1.0f / sum;
    bf16* prow = &P[tid * PROW];
    for (int c2 = 0; c2 < NKEY; ++c2) prow[c2] = f2bf(srow[c2] * inv);
    for (int c2 = NKEY; c2 < KPAD; ++c2) prow[c2] = bz;   // zero K padding
  }
  __syncthreads();

  // --- O = P(64x224) @ SVt^T(224x64), 4 N-tiles, 7 wmma K-steps ---
  v8f oacc[4];
#pragma unroll
  for (int t2 = 0; t2 < 4; ++t2) oacc[t2] = zero8();
  for (int k7 = 0; k7 < 7; ++k7) {
    int ks = k7 * 32;
    v16bf a = ldfragA(&P[(m0 + mrow) * PROW + ks + m16 * 8]);
#pragma unroll
    for (int t2 = 0; t2 < 4; ++t2) {
      v16bf bb = ldfragB(&SVt[(t2 * 16 + mrow) * PROW + ks + m16 * 8]);
      oacc[t2] = WMMA_BF16(a, bb, oacc[t2]);
    }
  }
  // store to [B, L, H] bf16 for the output projection GEMM
  bf16* Oh = O + ((size_t)b * L_SEQ + bi * 64) * DMODEL + h * DH;
#pragma unroll
  for (int t2 = 0; t2 < 4; ++t2)
#pragma unroll
    for (int r = 0; r < 8; ++r) {
      int row = m0 + r + m16 * 8;
      int dd = t2 * 16 + mrow;
      Oh[(size_t)row * DMODEL + dd] = f2bf(oacc[t2][r]);
    }
}

// ---------------------------------------------------------------------------
// Kernel 3: y = O @ Wo^T + bo  (f32 out)
// ---------------------------------------------------------------------------
__global__ __launch_bounds__(256) void oproj_kernel(
    const bf16* __restrict__ Ow, const bf16* __restrict__ W,
    const float* __restrict__ bo, float* __restrict__ out) {
  extern __shared__ char smem[];
  v8f acc[2][4];
  gemm_tile_128x128(Ow + (size_t)blockIdx.y * 128 * DMODEL,
                    W + (size_t)blockIdx.x * 128 * DMODEL, DMODEL,
                    (bf16*)smem, acc);
  const int tid  = threadIdx.x;
  const int lane = tid & 31, wave = tid >> 5;
  const int wm = wave >> 1, wn = wave & 1;
  const int m16 = lane >> 4, mrow = lane & 15;
#pragma unroll
  for (int ti = 0; ti < 2; ++ti)
#pragma unroll
    for (int tj = 0; tj < 4; ++tj)
#pragma unroll
      for (int r = 0; r < 8; ++r) {
        int row = blockIdx.y * 128 + wm * 32 + ti * 16 + r + m16 * 8;
        int col = blockIdx.x * 128 + wn * 64 + tj * 16 + mrow;
        out[(size_t)row * DMODEL + col] = acc[ti][tj][r] + bo[col];
      }
}

// ---------------------------------------------------------------------------
// Host launch
// ---------------------------------------------------------------------------
extern "C" void kernel_launch(void* const* d_in, const int* in_sizes, int n_in,
                              void* d_out, int out_size, void* d_ws, size_t ws_size,
                              hipStream_t stream) {
  const float* u    = (const float*)d_in[0];
  const float* Wqkv = (const float*)d_in[1];
  const float* bqkv = (const float*)d_in[2];
  const float* Wo   = (const float*)d_in[3];
  const float* bo   = (const float*)d_in[4];
  const float* snk  = (const float*)d_in[5];
  float* out = (float*)d_out;

  char* ws = (char*)d_ws;
  size_t off = 0;
  auto carve = [&](size_t bytes) -> char* {
    char* p = ws + off;
    off += (bytes + 255) & ~(size_t)255;
    return p;
  };
  const size_t M = (size_t)BATCH * L_SEQ;                       // 8192
  bf16* u_bf    = (bf16*)carve(M * DMODEL * 2);                 // reused as Ow
  bf16* wqkv_bf = (bf16*)carve((size_t)NQKV * DMODEL * 2);
  bf16* wo_bf   = (bf16*)carve((size_t)DMODEL * DMODEL * 2);
  bf16* snk_bf  = (bf16*)carve((size_t)NH * NSINK * DH * 2);
  float* ropeC  = (float*)carve((size_t)L_SEQ * 32 * 4);
  float* ropeS  = (float*)carve((size_t)L_SEQ * 32 * 4);
  bf16* Qw      = (bf16*)carve(M * DMODEL * 2);
  bf16* Kw      = (bf16*)carve(M * DMODEL * 2);
  bf16* Vw      = (bf16*)carve(M * DMODEL * 2);
  bf16* Ow      = u_bf;   // u_bf is dead after the QKV GEMM — alias it

  const int n_u = (int)(M * DMODEL);
  const int n_wqkv = NQKV * DMODEL;
  const int n_wo = DMODEL * DMODEL;
  const int n_snk = NH * NSINK * DH;
  cvt_kernel<<<(n_u + 255) / 256, 256, 0, stream>>>(u, u_bf, n_u);
  cvt_kernel<<<(n_wqkv + 255) / 256, 256, 0, stream>>>(Wqkv, wqkv_bf, n_wqkv);
  cvt_kernel<<<(n_wo + 255) / 256, 256, 0, stream>>>(Wo, wo_bf, n_wo);
  cvt_kernel<<<(n_snk + 255) / 256, 256, 0, stream>>>(snk, snk_bf, n_snk);
  rope_kernel<<<(L_SEQ * 32) / 256, 256, 0, stream>>>(ropeC, ropeS);

  (void)hipFuncSetAttribute((const void*)qkv_rope_kernel,
                            hipFuncAttributeMaxDynamicSharedMemorySize, GEMM_SMEM);
  qkv_rope_kernel<<<dim3(NQKV / 128, (int)(M / 128)), 256, GEMM_SMEM, stream>>>(
      u_bf, wqkv_bf, bqkv, ropeC, ropeS, Qw, Kw, Vw);

  const size_t attn_smem =
      (size_t)(64 * LDT + NKEY * LDT + 64 * PROW + 64 * PROW) * 2 +
      (size_t)64 * SROW * 4;                                    // 152832 B
  (void)hipFuncSetAttribute((const void*)attn_kernel,
                            hipFuncAttributeMaxDynamicSharedMemorySize, (int)attn_smem);
  attn_kernel<<<BATCH * NH * NB, 128, attn_smem, stream>>>(Qw, Kw, Vw, snk_bf, Ow);

  (void)hipFuncSetAttribute((const void*)oproj_kernel,
                            hipFuncAttributeMaxDynamicSharedMemorySize, GEMM_SMEM);
  oproj_kernel<<<dim3(DMODEL / 128, (int)(M / 128)), 256, GEMM_SMEM, stream>>>(
      Ow, wo_bf, bo, out);
}
